// RBFInterpolant_65927747994079
// MI455X (gfx1250) — matrix operand order repaired and checked by reading, not compiled, not run on hardware
//
#include <hip/hip_runtime.h>

typedef __attribute__((ext_vector_type(16))) __bf16 v16bf;
typedef __attribute__((ext_vector_type(8)))  __bf16 v8bf;
typedef __attribute__((ext_vector_type(4)))  __bf16 v4bf;
typedef __attribute__((ext_vector_type(8)))  float  v8f;

#define BM 128          // rows of x per workgroup
#define BN 128          // centers per LDS chunk
#define DD 64           // feature dim
#define LDSTR 72        // padded bf16 stride per center row (64 + 8) -> conflict-free b128

static __device__ __forceinline__ __bf16 f32_to_bf16_rne(float f) {
  unsigned u = __builtin_bit_cast(unsigned, f);
  u += 0x7FFFu + ((u >> 16) & 1u);
  unsigned short s = (unsigned short)(u >> 16);
  return __builtin_bit_cast(__bf16, s);
}
static __device__ __forceinline__ float bf16_to_f32(__bf16 b) {
  unsigned short s = __builtin_bit_cast(unsigned short, b);
  unsigned u = ((unsigned)s) << 16;
  return __builtin_bit_cast(float, u);
}

__global__ __launch_bounds__(256) void rbf_wmma_kernel(
    const float* __restrict__ x, const float* __restrict__ centers,
    const float* __restrict__ coefs, float* __restrict__ out, int M) {
  // Double-buffered center chunk: bf16 hi/lo tiles + f32 norms + coefs (~76 KB total)
  __shared__ __align__(16) __bf16 sh_hi[2][BN * LDSTR];
  __shared__ __align__(16) __bf16 sh_lo[2][BN * LDSTR];
  __shared__ float sh_cn[2][BN];
  __shared__ float sh_cf[2][BN];

  const int t      = threadIdx.x;
  const int lane   = t & 31;
  const int wave   = t >> 5;       // 0..7 -> 16-row sub-tile
  const int laneHi = lane >> 4;    // 0/1 : which K-half this lane holds
  const int laneLo = lane & 15;
  const int c      = t >> 1;       // staging: center within chunk
  const int h      = t & 1;        // staging: which 32-float half of the row

  // ---- Build this wave's 16x64 A tile (x rows) as bf16 hi/lo fragments.
  // ISA 16-bit A 16x32 layout: lane holds K = 8*laneHi + {0..7} and +16..+23 (elements 0..7 / 8..15).
  const int row = blockIdx.x * BM + wave * 16 + laneLo;
  const float* xr = x + (size_t)row * DD;
  v16bf Ahi[2], Alo[2];
  float pn = 0.f;
#pragma unroll
  for (int s = 0; s < 2; ++s) {
#pragma unroll
    for (int g = 0; g < 2; ++g) {
      const float4* p = (const float4*)(xr + 32 * s + 8 * laneHi + 16 * g);
      float4 v0 = p[0];
      float4 v1 = p[1];
      float f[8] = {v0.x, v0.y, v0.z, v0.w, v1.x, v1.y, v1.z, v1.w};
#pragma unroll
      for (int e = 0; e < 8; ++e) {
        pn = fmaf(f[e], f[e], pn);
        __bf16 hh = f32_to_bf16_rne(f[e]);
        __bf16 ll = f32_to_bf16_rne(f[e] - bf16_to_f32(hh));
        Ahi[s][8 * g + e] = hh;
        Alo[s][8 * g + e] = ll;
      }
    }
  }
  // Exact f32 row norm: lane L and L+16 hold disjoint K-halves.
  float xnorm_row = pn + __shfl_xor(pn, 16, 32);
  // D-matrix vgpr j maps to row (j + 8*laneHi): keep those 8 norms in registers.
  float xn8[8];
#pragma unroll
  for (int j = 0; j < 8; ++j) xn8[j] = __shfl(xnorm_row, j + 8 * laneHi, 16);

  float acc[8];
#pragma unroll
  for (int j = 0; j < 8; ++j) acc[j] = 0.f;

  // ---- Pipelined staging state: next chunk's raw f32 centers + coef live in registers.
  float4 rv[8];
  float  rcf = 0.f;

  auto load_regs = [&](int cb) {
    const float* src = centers + (size_t)(cb + c) * DD + h * 32;
#pragma unroll
    for (int i = 0; i < 8; ++i) rv[i] = ((const float4*)src)[i];
    if (t < BN) rcf = coefs[cb + t];
    // keep the stream warm: prefetch the chunk after next (global_prefetch_b8)
    if (cb + 2 * BN < M) __builtin_prefetch(src + (size_t)2 * BN * DD, 0, 0);
  };

  auto store_lds = [&](int buf) {
    float cn = 0.f;
#pragma unroll
    for (int i = 0; i < 8; ++i) {
      float f[4] = {rv[i].x, rv[i].y, rv[i].z, rv[i].w};
      v4bf hi4, lo4;
#pragma unroll
      for (int e = 0; e < 4; ++e) {
        cn = fmaf(f[e], f[e], cn);
        __bf16 hh = f32_to_bf16_rne(f[e]);
        hi4[e] = hh;
        lo4[e] = f32_to_bf16_rne(f[e] - bf16_to_f32(hh));
      }
      const int idx = c * LDSTR + h * 32 + i * 4;
      *(v4bf*)(&sh_hi[buf][idx]) = hi4;
      *(v4bf*)(&sh_lo[buf][idx]) = lo4;
    }
    float cn_full = cn + __shfl_xor(cn, 1, 32);
    if (h == 0) sh_cn[buf][c] = cn_full;
    if (t < BN) sh_cf[buf][t] = rcf;
  };

  const int nch = M / BN;
  load_regs(0);
  store_lds(0);

  for (int i = 0; i < nch; ++i) {
    const int buf = i & 1;
    if (i + 1 < nch) load_regs((i + 1) * BN);  // global latency overlapped with WMMAs below
    __syncthreads();  // buf's stores visible; other buffer's readers (iter i-1) done

    // ---- 8 column tiles of 16 centers; 6 bf16 WMMAs per tile (bf16x3 over K=64),
    //      split into two accumulator chains so the scheduler can overlap them.
#pragma unroll
    for (int ct = 0; ct < BN / 16; ++ct) {
      const int col = ct * 16 + laneLo;
      v8f S1 = {};  // hi*hi
      v8f S2 = {};  // hi*lo + lo*hi correction
#pragma unroll
      for (int s = 0; s < 2; ++s) {
        // B 32x16 layout: lane holds col=laneLo, K = 8*laneHi + {0..7} and +16..+23 (matches A).
        const __bf16* bh = &sh_hi[buf][col * LDSTR + 32 * s + 8 * laneHi];
        const __bf16* bl = &sh_lo[buf][col * LDSTR + 32 * s + 8 * laneHi];
        v8bf h0 = *(const v8bf*)(bh);
        v8bf h1 = *(const v8bf*)(bh + 16);
        v8bf l0 = *(const v8bf*)(bl);
        v8bf l1 = *(const v8bf*)(bl + 16);
        v16bf Bhi = __builtin_shufflevector(h0, h1, 0,1,2,3,4,5,6,7,8,9,10,11,12,13,14,15);
        v16bf Blo = __builtin_shufflevector(l0, l1, 0,1,2,3,4,5,6,7,8,9,10,11,12,13,14,15);
        S2 = __builtin_amdgcn_wmma_f32_16x16x32_bf16(false, Alo[s], false, Bhi, (short)0, S2, false, false);
        S2 = __builtin_amdgcn_wmma_f32_16x16x32_bf16(false, Ahi[s], false, Blo, (short)0, S2, false, false);
        S1 = __builtin_amdgcn_wmma_f32_16x16x32_bf16(false, Ahi[s], false, Bhi, (short)0, S1, false, false);
      }
      const float cn = sh_cn[buf][col];
      const float cf = sh_cf[buf][col];
#pragma unroll
      for (int j = 0; j < 8; ++j) {
        float S = S1[j] + S2[j];
        float r2 = fmaxf(fmaf(-2.f, S, xn8[j] + cn), 0.f);
        acc[j] = fmaf(cf * r2, __builtin_sqrtf(r2), acc[j]);  // coef * r^3
      }
    }

    if (i + 1 < nch) store_lds(1 - buf);  // other buffer is free after the barrier
  }

  // ---- Reduce each row over its 16 column-lanes (butterfly stays within half-wave).
#pragma unroll
  for (int j = 0; j < 8; ++j) {
#pragma unroll
    for (int m = 1; m < 16; m <<= 1) acc[j] += __shfl_xor(acc[j], m, 32);
  }
  const int rowbase = blockIdx.x * BM + wave * 16 + 8 * laneHi;
  if (laneLo == 0) {
#pragma unroll
    for (int j = 0; j < 8; ++j) out[rowbase + j] = acc[j];
  }
}

extern "C" void kernel_launch(void* const* d_in, const int* in_sizes, int n_in,
                              void* d_out, int out_size, void* d_ws, size_t ws_size,
                              hipStream_t stream) {
  const float* x       = (const float*)d_in[0];
  const float* centers = (const float*)d_in[1];
  const float* coefs   = (const float*)d_in[2];
  float* out = (float*)d_out;
  const int N = in_sizes[0] / DD;   // 32768
  const int M = in_sizes[1] / DD;   // 8192
  dim3 grid(N / BM);                // 256 workgroups of 8 waves
  rbf_wmma_kernel<<<grid, 256, 0, stream>>>(x, centers, coefs, out, M);
  (void)d_ws; (void)ws_size; (void)n_in; (void)out_size;
}
